// CustomLSTMCell_33328946217790
// MI455X (gfx1250) — matrix operand order repaired and checked by reading, not compiled, not run on hardware
//
#include <hip/hip_runtime.h>
#include <cstdint>
#include <cstddef>

// ---------------------------------------------------------------------------
// LSTM cell for gfx1250 (MI455X): bf16 WMMA, f32 accumulate.
//   z = [h_tm1 | x] @ W   (2048 x 1536) @ (1536 x 4096), fused gate epilogue.
// Pipeline: (1) cvt_A packs [h|x] -> bf16, (2) cvt_Wt converts+transposes W,
//           (3) lstm_gemm_bf16 runs WMMA GEMM + gates. Fallback kernel if ws
//           is too small does in-loop conversion (round-1 path).
// ---------------------------------------------------------------------------

typedef __attribute__((ext_vector_type(16))) __bf16 v16bf;
typedef __attribute__((ext_vector_type(8)))  float  v8f;

union Frag16 { v16bf v; uint4 q[2]; };

__device__ __forceinline__ unsigned short f2bf(float f) {
  unsigned int u = __builtin_bit_cast(unsigned int, f);
  u += 0x7FFFu + ((u >> 16) & 1u);          // round-to-nearest-even
  return (unsigned short)(u >> 16);
}
__device__ __forceinline__ float fsigmoid(float x) {
  return 1.0f / (1.0f + __expf(-x));
}
__device__ __forceinline__ float ftanh(float x) {
  return 1.0f - 2.0f / (__expf(2.0f * x) + 1.0f);
}

#define BATCH 2048
#define UNITS 1024
#define IDIM  512
#define KDIM  1536   // UNITS + IDIM
#define NDIM  4096   // 4 * UNITS
#define LDT   40     // padded ushort stride for LDS bf16 tiles (80B rows)
#define LDZ   132    // padded float stride for z staging

// ========================= prep kernel 1: A pack ===========================
// A_bf[m][k] = bf16( k < 1024 ? h[m][k] : x[m][k-1024] ), 4 elems/thread.
__global__ __launch_bounds__(256)
void cvt_A(const float* __restrict__ x, const float* __restrict__ h,
           unsigned short* __restrict__ Abf) {
  const size_t e = ((size_t)blockIdx.x * 256 + threadIdx.x) * 4;
  const int m = (int)(e / KDIM);
  const int k = (int)(e % KDIM);
  const float4 v = (k < UNITS)
      ? *(const float4*)&h[(size_t)m * UNITS + k]
      : *(const float4*)&x[(size_t)m * IDIM + (k - UNITS)];
  ushort4 b;
  b.x = f2bf(v.x); b.y = f2bf(v.y); b.z = f2bf(v.z); b.w = f2bf(v.w);
  *(ushort4*)&Abf[e] = b;
}

// ==================== prep kernel 2: W convert+transpose ===================
// Wt[n][k] = bf16(W[k][n]).  Tile 128(K) x 32(N) via LDS, coalesced both ways.
__global__ __launch_bounds__(256)
void cvt_Wt(const float* __restrict__ W, unsigned short* __restrict__ Wt) {
  __shared__ __align__(16) unsigned short T[32][136];  // [n][k], padded rows
  const int k0 = blockIdx.x * 128;
  const int n0 = blockIdx.y * 32;
  const int tid = threadIdx.x;
  #pragma unroll
  for (int p = 0; p < 4; ++p) {
    const int k  = (tid >> 3) + p * 32;
    const int n4 = (tid & 7) * 4;
    const float4 v = *(const float4*)&W[(size_t)(k0 + k) * NDIM + n0 + n4];
    T[n4 + 0][k] = f2bf(v.x);
    T[n4 + 1][k] = f2bf(v.y);
    T[n4 + 2][k] = f2bf(v.z);
    T[n4 + 3][k] = f2bf(v.w);
  }
  __syncthreads();
  #pragma unroll
  for (int p = 0; p < 2; ++p) {
    const int n  = (tid >> 4) + p * 16;
    const int kc = (tid & 15) * 8;
    *(uint4*)&Wt[(size_t)(n0 + n) * KDIM + k0 + kc] = *(const uint4*)&T[n][kc];
  }
}

// ===================== main kernel: WMMA GEMM + gates ======================
// Block: 128 batch rows x 32 units; N-tile = 4 gates x 32 units gathered.
__global__ __launch_bounds__(256)
void lstm_gemm_bf16(const unsigned short* __restrict__ Abf,  // [2048][1536]
                    const unsigned short* __restrict__ Wt,   // [4096][1536]
                    const float* __restrict__ c_tm1,
                    float* __restrict__ out) {
  __shared__ __align__(16) unsigned short As[128 * LDT];
  __shared__ __align__(16) unsigned short Bs[128 * LDT];
  __shared__ float zs[128 * LDZ];

  const int tid  = threadIdx.x;
  const int lane = tid & 31;
  const int w    = tid >> 5;
  const int m0w  = (w & 3) * 32;
  const int n0w  = (w >> 2) * 64;
  const int mblk = blockIdx.x * 128;
  const int u0   = blockIdx.y * 32;

  v8f acc[2][4];
  #pragma unroll
  for (int mt = 0; mt < 2; ++mt)
    #pragma unroll
    for (int nt = 0; nt < 4; ++nt)
      acc[mt][nt] = v8f{};

  // Staging maps (pure b128 copies, no per-element work)
  const int a_r  = tid >> 2;                   // 0..63, 2 passes
  const int a_kc = (tid & 3) * 8;              // 0,8,16,24
  const int b_np = tid >> 1;                   // 0..127 (n' row)
  const int b_g  = b_np >> 5;                  // gate 0..3
  const size_t b_row = (size_t)(b_g * UNITS + u0 + (b_np & 31)) * KDIM;

  const int kh = lane >> 4;
  const int lr = lane & 15;

  for (int kb = 0; kb < KDIM; kb += 32) {
    // ---- Stage A tile: [128 m][32 k] bf16, b128 copies --------------------
    #pragma unroll
    for (int p = 0; p < 2; ++p) {
      const int r = a_r + p * 64;
      const uint4 v = *(const uint4*)&Abf[(size_t)(mblk + r) * KDIM + kb + a_kc];
      *(uint4*)&As[r * LDT + a_kc] = v;
    }
    // ---- Stage B tile: [128 n'][32 k] bf16 (Wt already transposed) --------
    #pragma unroll
    for (int p = 0; p < 2; ++p) {
      const int kc = ((tid & 1) + p * 2) * 8;
      const uint4 v = *(const uint4*)&Wt[b_row + kb + kc];
      *(uint4*)&Bs[b_np * LDT + kc] = v;
    }
    // Prefetch next k-step (global_prefetch_b8)
    const int kn = kb + 32;
    if (kn < KDIM) {
      __builtin_prefetch(&Abf[(size_t)(mblk + a_r) * KDIM + kn + a_kc], 0, 1);
      __builtin_prefetch(&Wt[b_row + kn], 0, 1);
    }
    __syncthreads();

    // ---- Fragments (ISA 16-bit A 16x32 / B 32x16 layouts) -----------------
    Frag16 af[2];
    #pragma unroll
    for (int mt = 0; mt < 2; ++mt) {
      const int row = m0w + mt * 16 + lr;
      af[mt].q[0] = *(const uint4*)&As[row * LDT + kh * 8];
      af[mt].q[1] = *(const uint4*)&As[row * LDT + 16 + kh * 8];
    }
    Frag16 bf[4];
    #pragma unroll
    for (int nt = 0; nt < 4; ++nt) {
      const int col = n0w + nt * 16 + lr;
      bf[nt].q[0] = *(const uint4*)&Bs[col * LDT + kh * 8];
      bf[nt].q[1] = *(const uint4*)&Bs[col * LDT + 16 + kh * 8];
    }

    #pragma unroll
    for (int mt = 0; mt < 2; ++mt)
      #pragma unroll
      for (int nt = 0; nt < 4; ++nt)
        acc[mt][nt] = __builtin_amdgcn_wmma_f32_16x16x32_bf16(
            false, af[mt].v, false, bf[nt].v,
            (short)0, acc[mt][nt], false, false);

    __syncthreads();
  }

  // ---- Spill accumulators to LDS (documented C/D layout) -------------------
  #pragma unroll
  for (int mt = 0; mt < 2; ++mt)
    #pragma unroll
    for (int nt = 0; nt < 4; ++nt) {
      const int mrow = m0w + mt * 16 + 8 * kh;
      const int ncol = n0w + nt * 16 + lr;
      #pragma unroll
      for (int r = 0; r < 8; ++r)
        zs[(mrow + r) * LDZ + ncol] = acc[mt][nt][r];
    }
  __syncthreads();

  // ---- Fused LSTM gate epilogue -------------------------------------------
  float* __restrict__ out_h = out;
  float* __restrict__ out_c = out + (size_t)BATCH * UNITS;
  #pragma unroll
  for (int p = 0; p < 16; ++p) {
    const int idx = tid + p * 256;
    const int m = idx >> 5;
    const int u = idx & 31;
    const float z_i = zs[m * LDZ + u];
    const float z_f = zs[m * LDZ + 32 + u];
    const float z_c = zs[m * LDZ + 64 + u];
    const float z_o = zs[m * LDZ + 96 + u];
    const size_t row = (size_t)(mblk + m);
    const float cp = c_tm1[row * UNITS + u0 + u];

    const float ig = fsigmoid(z_i);
    const float fg = fsigmoid(z_f);
    const float ct = ftanh(z_c);
    const float og = fsigmoid(z_o);
    const float cn = fg * cp + ig * ct;
    const float hn = og * ftanh(cn);

    out_h[row * UNITS + u0 + u] = hn;
    out_c[row * UNITS + u0 + u] = cn;
  }
}

// ================= fallback: round-1 fused-conversion kernel ===============
__global__ __launch_bounds__(256)
void lstm_cell_wmma_fused(const float* __restrict__ x,
                          const float* __restrict__ h_tm1,
                          const float* __restrict__ c_tm1,
                          const float* __restrict__ Wk,
                          float* __restrict__ out) {
  __shared__ __align__(16) unsigned short As[128 * LDT];
  __shared__ __align__(16) unsigned short Bs[128 * LDT];
  __shared__ float zs[128 * LDZ];

  const int tid  = threadIdx.x;
  const int lane = tid & 31;
  const int w    = tid >> 5;
  const int m0w  = (w & 3) * 32;
  const int n0w  = (w >> 2) * 64;
  const int mblk = blockIdx.x * 128;
  const int u0   = blockIdx.y * 32;

  v8f acc[2][4];
  #pragma unroll
  for (int mt = 0; mt < 2; ++mt)
    #pragma unroll
    for (int nt = 0; nt < 4; ++nt)
      acc[mt][nt] = v8f{};

  const int a_c4 = (tid & 7) * 4;
  const int a_r  = tid >> 3;
  const int b_nn = (tid & 31) * 4;
  const int b_g  = b_nn >> 5;
  const int b_j  = b_nn & 31;
  const int b_kr = tid >> 5;
  const int kh = lane >> 4;
  const int lr = lane & 15;

  for (int kb = 0; kb < KDIM; kb += 32) {
    const float* srcA; int ldA, kofs;
    if (kb < UNITS) { srcA = h_tm1; ldA = UNITS; kofs = kb; }
    else            { srcA = x;     ldA = IDIM;  kofs = kb - UNITS; }

    #pragma unroll
    for (int p = 0; p < 4; ++p) {
      const int r = a_r + p * 32;
      const float4 v = *(const float4*)&srcA[(size_t)(mblk + r) * ldA + kofs + a_c4];
      ushort4 bv;
      bv.x = f2bf(v.x); bv.y = f2bf(v.y); bv.z = f2bf(v.z); bv.w = f2bf(v.w);
      *(ushort4*)&As[r * LDT + a_c4] = bv;
    }
    const float* srcB = &Wk[(size_t)kb * NDIM + (size_t)b_g * UNITS + u0 + b_j];
    #pragma unroll
    for (int p = 0; p < 4; ++p) {
      const int kr = b_kr + p * 8;
      const float4 v = *(const float4*)&srcB[(size_t)kr * NDIM];
      Bs[(b_nn + 0) * LDT + kr] = f2bf(v.x);
      Bs[(b_nn + 1) * LDT + kr] = f2bf(v.y);
      Bs[(b_nn + 2) * LDT + kr] = f2bf(v.z);
      Bs[(b_nn + 3) * LDT + kr] = f2bf(v.w);
    }
    __syncthreads();

    Frag16 af[2];
    #pragma unroll
    for (int mt = 0; mt < 2; ++mt) {
      const int row = m0w + mt * 16 + lr;
      af[mt].q[0] = *(const uint4*)&As[row * LDT + kh * 8];
      af[mt].q[1] = *(const uint4*)&As[row * LDT + 16 + kh * 8];
    }
    Frag16 bf[4];
    #pragma unroll
    for (int nt = 0; nt < 4; ++nt) {
      const int col = n0w + nt * 16 + lr;
      bf[nt].q[0] = *(const uint4*)&Bs[col * LDT + kh * 8];
      bf[nt].q[1] = *(const uint4*)&Bs[col * LDT + 16 + kh * 8];
    }
    #pragma unroll
    for (int mt = 0; mt < 2; ++mt)
      #pragma unroll
      for (int nt = 0; nt < 4; ++nt)
        acc[mt][nt] = __builtin_amdgcn_wmma_f32_16x16x32_bf16(
            false, af[mt].v, false, bf[nt].v,
            (short)0, acc[mt][nt], false, false);
    __syncthreads();
  }

  #pragma unroll
  for (int mt = 0; mt < 2; ++mt)
    #pragma unroll
    for (int nt = 0; nt < 4; ++nt) {
      const int mrow = m0w + mt * 16 + 8 * kh;
      const int ncol = n0w + nt * 16 + lr;
      #pragma unroll
      for (int r = 0; r < 8; ++r)
        zs[(mrow + r) * LDZ + ncol] = acc[mt][nt][r];
    }
  __syncthreads();

  float* __restrict__ out_h = out;
  float* __restrict__ out_c = out + (size_t)BATCH * UNITS;
  #pragma unroll
  for (int p = 0; p < 16; ++p) {
    const int idx = tid + p * 256;
    const int m = idx >> 5;
    const int u = idx & 31;
    const float z_i = zs[m * LDZ + u];
    const float z_f = zs[m * LDZ + 32 + u];
    const float z_c = zs[m * LDZ + 64 + u];
    const float z_o = zs[m * LDZ + 96 + u];
    const size_t row = (size_t)(mblk + m);
    const float cp = c_tm1[row * UNITS + u0 + u];
    const float ig = fsigmoid(z_i);
    const float fg = fsigmoid(z_f);
    const float ct = ftanh(z_c);
    const float og = fsigmoid(z_o);
    const float cn = fg * cp + ig * ct;
    const float hn = og * ftanh(cn);
    out_h[row * UNITS + u0 + u] = hn;
    out_c[row * UNITS + u0 + u] = cn;
  }
}

// ============================== launcher ===================================
extern "C" void kernel_launch(void* const* d_in, const int* in_sizes, int n_in,
                              void* d_out, int out_size, void* d_ws, size_t ws_size,
                              hipStream_t stream) {
  (void)in_sizes; (void)n_in; (void)out_size;
  const float* x     = (const float*)d_in[0];  // [2048, 512]
  const float* h_tm1 = (const float*)d_in[1];  // [2048, 1024]
  const float* c_tm1 = (const float*)d_in[2];  // [2048, 1024]
  const float* Wk    = (const float*)d_in[3];  // [1536, 4096]
  float* out         = (float*)d_out;          // h then c, each 2048*1024 f32

  const size_t needA = (size_t)BATCH * KDIM * sizeof(unsigned short); //  6 MB
  const size_t needW = (size_t)NDIM  * KDIM * sizeof(unsigned short); // 12 MB

  if (ws_size >= needA + needW && d_ws != nullptr) {
    unsigned short* Abf = (unsigned short*)d_ws;
    unsigned short* Wt  = (unsigned short*)((char*)d_ws + needA);

    cvt_A<<<(BATCH * KDIM) / (256 * 4), 256, 0, stream>>>(x, h_tm1, Abf);
    cvt_Wt<<<dim3(KDIM / 128, NDIM / 32), 256, 0, stream>>>(Wk, Wt);
    lstm_gemm_bf16<<<dim3(BATCH / 128, UNITS / 32), 256, 0, stream>>>(
        Abf, Wt, c_tm1, out);
  } else {
    lstm_cell_wmma_fused<<<dim3(BATCH / 128, UNITS / 32), 256, 0, stream>>>(
        x, h_tm1, c_tm1, Wk, out);
  }
}
